// BertRGCN_48507360641519
// MI455X (gfx1250) — matrix operand order repaired and checked by reading, not compiled, not run on hardware
//
#include <hip/hip_runtime.h>

#define N_NODES 50000
#define B_G     100
#define L_SEQ   500
#define LPAD    512
#define E_EDGES 300000
#define EMB     768
#define DMODEL  256
#define NREL    3
#define NLAYERS 6
#define NHEADS  4
#define DH      64
#define FFN_DIM 1024

typedef __attribute__((ext_vector_type(16))) __bf16        v16bf;
typedef __attribute__((ext_vector_type(8)))  float         v8f;
typedef __attribute__((ext_vector_type(16))) unsigned short u16x16;

__device__ __forceinline__ unsigned short f32_to_bf16(float f) {
  unsigned int u = __float_as_uint(f);
  unsigned int lsb = (u >> 16) & 1u;
  u += 0x7fffu + lsb;               // round-to-nearest-even
  return (unsigned short)(u >> 16);
}
__device__ __forceinline__ int imin_(int a, int b) { return a < b ? a : b; }

// ---------------------------------------------------------------- cast f32->bf16
__global__ __launch_bounds__(256)
void castf32_bf16_kernel(const float* __restrict__ in, unsigned short* __restrict__ out, int n) {
  int i = blockIdx.x * 256 + threadIdx.x;
  if (i < n) out[i] = f32_to_bf16(in[i]);
}

// ---------------------------------------------------------------- WMMA GEMM
// C[M,N] = A[M,K] @ B[K,N] (+bias) (relu?), A f32 (converted on the fly) or bf16,
// B bf16 (pre-cast). Outputs: optional f32 C, optional bf16 C.
#define BM 64
#define BN 64
#define BK 32

__global__ __launch_bounds__(128)
void gemm_wmma(const void* __restrict__ Aptr, int a_is_f32,
               const unsigned short* __restrict__ Bw,
               const float* __restrict__ bias,
               float* __restrict__ Cf, unsigned short* __restrict__ Cb,
               int M, int N, int K, int relu)
{
  __shared__ unsigned short As[BM][BK + 8];   // row-major A tile (bf16 bits)
  __shared__ unsigned short Bs[BN][BK + 8];   // transposed B tile: Bs[n][k]
  const int tid  = threadIdx.x;
  const int lane = tid & 31;
  const int wid  = tid >> 5;                  // 4 waves
  const int wrow = wid >> 1, wcol = wid & 1;  // 2x2 wave grid, 32x32 per wave
  const int m0 = blockIdx.y * BM;
  const int n0 = blockIdx.x * BN;
  const int hh = lane >> 4, l15 = lane & 15;

  v8f acc[2][2];
  acc[0][0] = {}; acc[0][1] = {}; acc[1][0] = {}; acc[1][1] = {};

  for (int kt = 0; kt < K; kt += BK) {
    if (kt + BK < K)  // warm L2/WGP$ for the next weight tile
      __builtin_prefetch((const char*)Bw + ((size_t)(kt + BK) * N + n0) * 2, 0, 1);

    // fill A tile: 64x32, 16 elements per thread
    for (int c = tid; c < (BM * BK / 8); c += 128) {
      int r = c >> 2, cc = (c & 3) * 8;
      int gr = m0 + r;
      unsigned short tmp[8];
      if (gr < M) {
        if (a_is_f32) {
          const float* Af = (const float*)Aptr + (size_t)gr * K + (kt + cc);
          #pragma unroll
          for (int i = 0; i < 8; ++i) tmp[i] = f32_to_bf16(Af[i]);
        } else {
          const unsigned short* Ah = (const unsigned short*)Aptr + (size_t)gr * K + (kt + cc);
          #pragma unroll
          for (int i = 0; i < 8; ++i) tmp[i] = Ah[i];
        }
      } else {
        #pragma unroll
        for (int i = 0; i < 8; ++i) tmp[i] = 0;
      }
      #pragma unroll
      for (int i = 0; i < 8; ++i) As[r][cc + i] = tmp[i];
    }
    // fill B tile transposed: Bs[n][k] = B[kt+k][n0+n]
    for (int c = tid; c < (BK * BN / 8); c += 128) {
      int k = c >> 3, nc = (c & 7) * 8;
      const unsigned short* Bp = Bw + (size_t)(kt + k) * N + (n0 + nc);
      #pragma unroll
      for (int i = 0; i < 8; ++i) Bs[nc + i][k] = Bp[i];
    }
    __syncthreads();

    // fragment loads: contiguous k = 16*hh + j for both A(16x32) and B(32x16)
    v16bf afrag[2], bfrag[2];
    #pragma unroll
    for (int fm = 0; fm < 2; ++fm) {
      int row = wrow * 32 + fm * 16 + l15;
      u16x16 ua;
      #pragma unroll
      for (int j = 0; j < 16; ++j) {
        int kk = hh * 16 + j;
        ua[j] = As[row][kk];
      }
      afrag[fm] = __builtin_bit_cast(v16bf, ua);
    }
    #pragma unroll
    for (int fn = 0; fn < 2; ++fn) {
      int col = wcol * 32 + fn * 16 + l15;
      u16x16 ub;
      #pragma unroll
      for (int j = 0; j < 16; ++j) ub[j] = Bs[col][hh * 16 + j];
      bfrag[fn] = __builtin_bit_cast(v16bf, ub);
    }
    #pragma unroll
    for (int fm = 0; fm < 2; ++fm)
      #pragma unroll
      for (int fn = 0; fn < 2; ++fn)
        acc[fm][fn] = __builtin_amdgcn_wmma_f32_16x16x32_bf16(
            false, afrag[fm], false, bfrag[fn], (short)0, acc[fm][fn], false, false);
    __syncthreads();
  }

  // epilogue: D layout m = i + 8*(lane>>4), n = lane&15
  #pragma unroll
  for (int fm = 0; fm < 2; ++fm)
    #pragma unroll
    for (int fn = 0; fn < 2; ++fn) {
      int n = n0 + wcol * 32 + fn * 16 + l15;
      float bv = bias ? bias[n] : 0.0f;
      #pragma unroll
      for (int i = 0; i < 8; ++i) {
        int m = m0 + wrow * 32 + fm * 16 + i + hh * 8;
        if (m < M) {
          float v = acc[fm][fn][i] + bv;
          if (relu) v = fmaxf(v, 0.0f);
          if (Cf) Cf[(size_t)m * N + n] = v;
          if (Cb) Cb[(size_t)m * N + n] = f32_to_bf16(v);
        }
      }
    }
}

// ---------------------------------------------------------------- RGCN scatter pieces
__global__ __launch_bounds__(256)
void count_edges_kernel(const int* __restrict__ et, const int* __restrict__ dst,
                        float* __restrict__ cnt, int E, int Nn) {
  int e = blockIdx.x * 256 + threadIdx.x;
  if (e < E) unsafeAtomicAdd(&cnt[(size_t)et[e] * Nn + dst[e]], 1.0f);
}

__global__ __launch_bounds__(256)
void recip_kernel(float* __restrict__ c, int n) {
  int i = blockIdx.x * 256 + threadIdx.x;
  if (i < n) c[i] = 1.0f / fmaxf(c[i], 1.0f);
}

// out[dst] += tr[src] * rcnt[r][dst] for edges of relation r (64 threads/edge x 4 dims)
__global__ __launch_bounds__(256)
void scatter_rel_kernel(const float* __restrict__ tr, const int* __restrict__ src,
                        const int* __restrict__ dst, const int* __restrict__ et,
                        const float* __restrict__ rcnt, float* __restrict__ out,
                        int r, int E, int Nn) {
  int idx = blockIdx.x * 256 + threadIdx.x;
  int e = idx >> 6;
  int d0 = (idx & 63) * 4;
  if (e >= E || et[e] != r) return;
  int s = src[e], dt = dst[e];
  float w = rcnt[(size_t)r * Nn + dt];
  const float* tp = tr + (size_t)s * DMODEL + d0;
  float* op = out + (size_t)dt * DMODEL + d0;
  #pragma unroll
  for (int i = 0; i < 4; ++i) unsafeAtomicAdd(op + i, tp[i] * w);
}

__global__ __launch_bounds__(256)
void relu_kernel(const float* __restrict__ in, float* __restrict__ out, int n) {
  int i = blockIdx.x * 256 + threadIdx.x;
  if (i < n) out[i] = fmaxf(in[i], 0.0f);
}

// ---------------------------------------------------------------- attention (per b,head,16-q-tile)
__global__ __launch_bounds__(32)
void attn_wmma(const unsigned short* __restrict__ qb,
               const unsigned short* __restrict__ kb,
               const unsigned short* __restrict__ vb,
               unsigned short* __restrict__ aob)
{
  __shared__ float          sc[16 * LPAD];     // 32 KB score tile
  __shared__ unsigned short pb[16 * LPAD];     // 16 KB bf16 probs
  const int qt = blockIdx.x;                   // 0..31 (512/16)
  const int bh = blockIdx.y;
  const int b  = bh >> 2;
  const int h  = bh & (NHEADS - 1);
  const int lane = threadIdx.x;
  const int hh = lane >> 4, l15 = lane & 15;
  const float scale = 0.125f;                  // 1/sqrt(64)

  int qrow  = qt * 16 + l15;
  int qnode = b * L_SEQ + imin_(qrow, L_SEQ - 1);
  const unsigned short* qp = qb + (size_t)qnode * DMODEL + h * DH;
  v16bf aq[2];
  #pragma unroll
  for (int c = 0; c < 2; ++c) {
    u16x16 ua;
    #pragma unroll
    for (int j = 0; j < 16; ++j) {
      int kk = c * 32 + hh * 16 + j;
      ua[j] = qp[kk];
    }
    aq[c] = __builtin_bit_cast(v16bf, ua);
  }

  // Q @ K^T over all 512 (padded) keys
  for (int kt = 0; kt < LPAD / 16; ++kt) {
    int key   = kt * 16 + l15;
    int knode = b * L_SEQ + imin_(key, L_SEQ - 1);
    const unsigned short* kp = kb + (size_t)knode * DMODEL + h * DH;
    v8f s = {};
    #pragma unroll
    for (int c = 0; c < 2; ++c) {
      u16x16 ub;
      #pragma unroll
      for (int j = 0; j < 16; ++j) ub[j] = kp[c * 32 + hh * 16 + j];
      s = __builtin_amdgcn_wmma_f32_16x16x32_bf16(
          false, aq[c], false, __builtin_bit_cast(v16bf, ub), (short)0, s, false, false);
    }
    float colmask = (key < L_SEQ) ? 0.0f : -1e9f;
    #pragma unroll
    for (int i = 0; i < 8; ++i) {
      int m = i + hh * 8;
      sc[m * LPAD + key] = s[i] * scale + colmask;
    }
  }
  __syncthreads();

  // softmax: one row per thread (16 rows)
  if (lane < 16) {
    float mx = -1e30f;
    for (int j = 0; j < LPAD; ++j) mx = fmaxf(mx, sc[lane * LPAD + j]);
    float sum = 0.0f;
    for (int j = 0; j < LPAD; ++j) {
      float e = __expf(sc[lane * LPAD + j] - mx);
      sc[lane * LPAD + j] = e;
      sum += e;
    }
    float inv = 1.0f / sum;
    for (int j = 0; j < LPAD; ++j)
      pb[lane * LPAD + j] = f32_to_bf16(sc[lane * LPAD + j] * inv);
  }
  __syncthreads();

  // probs @ V  (keys contracted 32 at a time, 4 chunks of 16 output dims)
  v8f o[4];
  o[0] = {}; o[1] = {}; o[2] = {}; o[3] = {};
  for (int kt2 = 0; kt2 < LPAD / 32; ++kt2) {
    u16x16 ua;
    #pragma unroll
    for (int j = 0; j < 16; ++j) {
      int kk = kt2 * 32 + hh * 16 + j;
      ua[j] = pb[l15 * LPAD + kk];
    }
    v16bf af = __builtin_bit_cast(v16bf, ua);
    #pragma unroll
    for (int nc = 0; nc < 4; ++nc) {
      u16x16 ub;
      #pragma unroll
      for (int j = 0; j < 16; ++j) {
        int key   = kt2 * 32 + hh * 16 + j;
        int vnode = b * L_SEQ + imin_(key, L_SEQ - 1);   // padded keys have prob 0
        ub[j] = vb[(size_t)vnode * DMODEL + h * DH + nc * 16 + l15];
      }
      o[nc] = __builtin_amdgcn_wmma_f32_16x16x32_bf16(
          false, af, false, __builtin_bit_cast(v16bf, ub), (short)0, o[nc], false, false);
    }
  }
  #pragma unroll
  for (int nc = 0; nc < 4; ++nc)
    #pragma unroll
    for (int i = 0; i < 8; ++i) {
      int m = i + hh * 8;
      int qr = qt * 16 + m;
      if (qr < L_SEQ)
        aob[(size_t)(b * L_SEQ + qr) * DMODEL + h * DH + nc * 16 + l15] = f32_to_bf16(o[nc][i]);
    }
}

// ---------------------------------------------------------------- residual + layernorm (D=256)
__global__ __launch_bounds__(256)
void resid_ln_kernel(const float* __restrict__ x, const float* __restrict__ res,
                     const float* __restrict__ g, const float* __restrict__ bta,
                     float* __restrict__ out) {
  int row = blockIdx.x;
  int t = threadIdx.x;
  size_t base = (size_t)row * DMODEL;
  float v = x[base + t] + res[base + t];
  __shared__ float sh[256];
  sh[t] = v; __syncthreads();
  for (int s = 128; s > 0; s >>= 1) { if (t < s) sh[t] += sh[t + s]; __syncthreads(); }
  float mu = sh[0] * (1.0f / 256.0f);
  __syncthreads();
  float d = v - mu;
  sh[t] = d * d; __syncthreads();
  for (int s = 128; s > 0; s >>= 1) { if (t < s) sh[t] += sh[t + s]; __syncthreads(); }
  float var = sh[0] * (1.0f / 256.0f);
  out[base + t] = d * rsqrtf(var + 1e-5f) * g[t] + bta[t];
}

// ---------------------------------------------------------------- mean pool over L
__global__ __launch_bounds__(256)
void pool_kernel(const float* __restrict__ h2, float* __restrict__ pooled) {
  int b = blockIdx.x;
  int t = threadIdx.x;
  float s = 0.0f;
  for (int i = 0; i < L_SEQ; ++i) s += h2[((size_t)b * L_SEQ + i) * DMODEL + t];
  pooled[(size_t)b * DMODEL + t] = s * (1.0f / (float)L_SEQ);
}

// ---------------------------------------------------------------- classifier 256 -> 2
__global__ __launch_bounds__(64)
void cls_kernel(const float* __restrict__ gr, const float* __restrict__ w,
                const float* __restrict__ b, float* __restrict__ out) {
  int idx = blockIdx.x * 64 + threadIdx.x;
  if (idx >= B_G * 2) return;
  int bb = idx >> 1, c = idx & 1;
  float s = b[c];
  for (int k = 0; k < DMODEL; ++k) s += gr[(size_t)bb * DMODEL + k] * w[k * 2 + c];
  out[idx] = s;
}

// ================================================================ host
extern "C" void kernel_launch(void* const* d_in, const int* in_sizes, int n_in,
                              void* d_out, int out_size, void* d_ws, size_t ws_size,
                              hipStream_t stream) {
  (void)in_sizes; (void)n_in; (void)out_size; (void)ws_size;
  const float* x      = (const float*)d_in[0];
  const int*   eidx   = (const int*)d_in[1];
  const int*   etyp   = (const int*)d_in[2];
  const float* in_w   = (const float*)d_in[4];
  const float* in_b   = (const float*)d_in[5];
  const float* rel_w  = (const float*)d_in[6];
  const float* root_w = (const float*)d_in[7];
  const float* conv_b = (const float*)d_in[8];
  const float* code_w = (const float*)d_in[9];
  const float* code_b = (const float*)d_in[10];
  const float* graph_w= (const float*)d_in[11];
  const float* graph_b= (const float*)d_in[12];
  const float* q_w = (const float*)d_in[13]; const float* q_b = (const float*)d_in[14];
  const float* k_w = (const float*)d_in[15]; const float* k_b = (const float*)d_in[16];
  const float* v_w = (const float*)d_in[17]; const float* v_b = (const float*)d_in[18];
  const float* o_w = (const float*)d_in[19]; const float* o_b = (const float*)d_in[20];
  const float* ln1_g = (const float*)d_in[21]; const float* ln1_b = (const float*)d_in[22];
  const float* f1_w = (const float*)d_in[23]; const float* f1_b = (const float*)d_in[24];
  const float* f2_w = (const float*)d_in[25]; const float* f2_b = (const float*)d_in[26];
  const float* ln2_g = (const float*)d_in[27]; const float* ln2_b = (const float*)d_in[28];
  const float* fo_w = (const float*)d_in[29]; const float* fo_b = (const float*)d_in[30];
  const float* cls_w = (const float*)d_in[31]; const float* cls_b = (const float*)d_in[32];
  const int* esrc = eidx;
  const int* edst = eidx + E_EDGES;

  char* ws = (char*)d_ws;
  size_t off = 0;
  auto carve = [&](size_t bytes) -> void* {
    void* p = ws + off; off += (bytes + 255) & ~(size_t)255; return p;
  };
  const size_t NB4 = (size_t)N_NODES * DMODEL * sizeof(float);
  float* buf1 = (float*)carve(NB4);          // h / h1
  float* buf2 = (float*)carve(NB4);          // rgcn out / ao(bf16) / h2
  float* buf3 = (float*)carve(NB4);          // tr / g / oo / ff2
  float* buf4 = (float*)carve(NB4);          // c
  float* buf5 = (float*)carve(NB4);          // q(bf16)
  float* buf67 = (float*)carve(2 * NB4);     // k,v (bf16) then ff1 (bf16, N x 1024)
  float* rcnt = (float*)carve((size_t)NREL * N_NODES * sizeof(float));
  float* pooled = (float*)carve((size_t)B_G * DMODEL * sizeof(float));
  float* grepr  = (float*)carve((size_t)B_G * DMODEL * sizeof(float));

  unsigned short* w_in    = (unsigned short*)carve(2 * (size_t)EMB * DMODEL);
  unsigned short* w_rel   = (unsigned short*)carve(2 * (size_t)NLAYERS * NREL * DMODEL * DMODEL);
  unsigned short* w_root  = (unsigned short*)carve(2 * (size_t)NLAYERS * DMODEL * DMODEL);
  unsigned short* w_code  = (unsigned short*)carve(2 * (size_t)EMB * DMODEL);
  unsigned short* w_graph = (unsigned short*)carve(2 * (size_t)DMODEL * DMODEL);
  unsigned short* w_q     = (unsigned short*)carve(2 * (size_t)DMODEL * DMODEL);
  unsigned short* w_k     = (unsigned short*)carve(2 * (size_t)DMODEL * DMODEL);
  unsigned short* w_v     = (unsigned short*)carve(2 * (size_t)DMODEL * DMODEL);
  unsigned short* w_o     = (unsigned short*)carve(2 * (size_t)DMODEL * DMODEL);
  unsigned short* w_f1    = (unsigned short*)carve(2 * (size_t)DMODEL * FFN_DIM);
  unsigned short* w_f2    = (unsigned short*)carve(2 * (size_t)FFN_DIM * DMODEL);
  unsigned short* w_fo    = (unsigned short*)carve(2 * (size_t)DMODEL * DMODEL);

  unsigned short* qb  = (unsigned short*)buf5;
  unsigned short* kb  = (unsigned short*)buf67;
  unsigned short* vb  = ((unsigned short*)buf67) + (size_t)N_NODES * DMODEL;
  unsigned short* aob = (unsigned short*)buf2;
  unsigned short* ffb = (unsigned short*)buf67;   // N x 1024 bf16, spans both halves

  auto cast = [&](const float* s, unsigned short* d, size_t n) {
    castf32_bf16_kernel<<<(int)((n + 255) / 256), 256, 0, stream>>>(s, d, (int)n);
  };
  auto gemm = [&](const void* A, int a_f32, const unsigned short* Bw, const float* bias,
                  float* Cf, unsigned short* Cb, int M, int Nc, int K, int relu) {
    dim3 g(Nc / 64, (M + 63) / 64);
    gemm_wmma<<<g, 128, 0, stream>>>(A, a_f32, Bw, bias, Cf, Cb, M, Nc, K, relu);
  };

  // -- weights to bf16 (tiny, sits in L2 afterwards)
  cast(in_w,   w_in,   (size_t)EMB * DMODEL);
  cast(rel_w,  w_rel,  (size_t)NLAYERS * NREL * DMODEL * DMODEL);
  cast(root_w, w_root, (size_t)NLAYERS * DMODEL * DMODEL);
  cast(code_w, w_code, (size_t)EMB * DMODEL);
  cast(graph_w,w_graph,(size_t)DMODEL * DMODEL);
  cast(q_w, w_q, (size_t)DMODEL * DMODEL);
  cast(k_w, w_k, (size_t)DMODEL * DMODEL);
  cast(v_w, w_v, (size_t)DMODEL * DMODEL);
  cast(o_w, w_o, (size_t)DMODEL * DMODEL);
  cast(f1_w, w_f1, (size_t)DMODEL * FFN_DIM);
  cast(f2_w, w_f2, (size_t)FFN_DIM * DMODEL);
  cast(fo_w, w_fo, (size_t)DMODEL * DMODEL);

  // -- per-(relation,dst) edge counts -> reciprocals (constant across layers)
  hipMemsetAsync(rcnt, 0, (size_t)NREL * N_NODES * sizeof(float), stream);
  count_edges_kernel<<<(E_EDGES + 255) / 256, 256, 0, stream>>>(etyp, edst, rcnt, E_EDGES, N_NODES);
  recip_kernel<<<(NREL * N_NODES + 255) / 256, 256, 0, stream>>>(rcnt, NREL * N_NODES);

  // -- input projection: h0 = x @ in_w + in_b
  gemm(x, 1, w_in, in_b, buf1, nullptr, N_NODES, DMODEL, EMB, 0);

  // -- RGCN layers
  const int nelem = N_NODES * DMODEL;
  for (int l = 0; l < NLAYERS; ++l) {
    gemm(buf1, 1, w_root + (size_t)l * DMODEL * DMODEL, conv_b + (size_t)l * DMODEL,
         buf2, nullptr, N_NODES, DMODEL, DMODEL, 0);
    for (int r = 0; r < NREL; ++r) {
      gemm(buf1, 1, w_rel + (size_t)(l * NREL + r) * DMODEL * DMODEL, nullptr,
           buf3, nullptr, N_NODES, DMODEL, DMODEL, 0);
      scatter_rel_kernel<<<(E_EDGES * 64 + 255) / 256, 256, 0, stream>>>(
          buf3, esrc, edst, etyp, rcnt, buf2, r, E_EDGES, N_NODES);
    }
    relu_kernel<<<(nelem + 255) / 256, 256, 0, stream>>>(buf2, buf1, nelem);
  }

  // -- fusion projections
  gemm(x,    1, w_code,  code_b,  buf4, nullptr, N_NODES, DMODEL, EMB,    0);  // c
  gemm(buf1, 1, w_graph, graph_b, buf3, nullptr, N_NODES, DMODEL, DMODEL, 0);  // g
  gemm(buf4, 1, w_q, q_b, nullptr, qb, N_NODES, DMODEL, DMODEL, 0);
  gemm(buf3, 1, w_k, k_b, nullptr, kb, N_NODES, DMODEL, DMODEL, 0);
  gemm(buf3, 1, w_v, v_b, nullptr, vb, N_NODES, DMODEL, DMODEL, 0);

  // -- cross attention
  {
    dim3 g(LPAD / 16, B_G * NHEADS);
    attn_wmma<<<g, 32, 0, stream>>>(qb, kb, vb, aob);
  }

  // -- output proj, residual + LN1
  gemm(aob, 0, w_o, o_b, buf3, nullptr, N_NODES, DMODEL, DMODEL, 0);           // oo
  resid_ln_kernel<<<N_NODES, 256, 0, stream>>>(buf3, buf4, ln1_g, ln1_b, buf1); // h1

  // -- FFN
  gemm(buf1, 1, w_f1, f1_b, nullptr, ffb, N_NODES, FFN_DIM, DMODEL, 1);        // relu(h1@f1)
  gemm(ffb,  0, w_f2, f2_b, buf3, nullptr, N_NODES, DMODEL, FFN_DIM, 0);       // ff2
  resid_ln_kernel<<<N_NODES, 256, 0, stream>>>(buf3, buf1, ln2_g, ln2_b, buf2); // h2

  // -- pool, final projections
  pool_kernel<<<B_G, 256, 0, stream>>>(buf2, pooled);
  gemm(pooled, 1, w_fo, fo_b, grepr, nullptr, B_G, DMODEL, DMODEL, 0);
  cls_kernel<<<(B_G * 2 + 63) / 64, 64, 0, stream>>>(grepr, cls_w, cls_b, (float*)d_out);
}
